// LiquidNeuralCell_68813966017013
// MI455X (gfx1250) — matrix operand order repaired and checked by reading, not compiled, not run on hardware
//
#include <hip/hip_runtime.h>
#include <hip/hip_bf16.h>
#include <math.h>

typedef __attribute__((ext_vector_type(16))) __bf16 v16bf;
typedef __attribute__((ext_vector_type(8)))  __bf16 v8bf;
typedef __attribute__((ext_vector_type(8)))  float  v8f;
typedef __attribute__((ext_vector_type(4)))  float  v4f;
typedef __attribute__((ext_vector_type(4)))  int    v4i;
typedef __attribute__((address_space(1)))    v4i    gv4i;   // global v4i
typedef __attribute__((address_space(3)))    v4i    lv4i;   // LDS v4i

union V16 { v16bf v; v8bf h[2]; };

static constexpr int Bn = 64, Tn = 512, Dn = 2048, Un = 1024;
static constexpr int Mtot = Bn * Tn;              // 32768 rows of x / xw

// ---------------------------------------------------------------------------
// Phase 1: xw = x @ kernel + bias   (M=32768, K=2048, N=1024), bf16 WMMA.
// 512 threads (16 waves), WG tile 256x128, wave tile 32x64,
// double-buffered LDS: one barrier per K-stage, loads overlap WMMA.
// ---------------------------------------------------------------------------
static constexpr int TM = 256, TN = 128, TK = 32;
static constexpr int NKB = Dn / TK;               // 64 K-stages
static constexpr int LDA = 40;                    // padded LDS row stride (halfwords)

__global__ __launch_bounds__(512)
void xw_gemm(const float* __restrict__ X, const float* __restrict__ W,
             const float* __restrict__ bias, float* __restrict__ out) {
  __shared__ __align__(16) __bf16 As[2][TM * LDA];   // 2 x 20 KB, [m][k]
  __shared__ __align__(16) __bf16 Bs[2][TN * LDA];   // 2 x 10 KB, [n][k] transposed

  const int n0   = blockIdx.x * TN;
  const int m0   = blockIdx.y * TM;
  const int tid  = threadIdx.x;
  const int lane = tid & 31;
  const int wid  = tid >> 5;                      // 16 waves: 8 (M) x 2 (N)
  const int wm   = (wid & 7) * 32;                // wave M offset
  const int wn   = (wid >> 3) * 64;               // wave N offset
  const int nl   = lane & 15;
  const int hid  = lane >> 4;

  v8f acc[2][4] = {};

  const int a_c4 = tid & 7;                       // float4 col in 32-wide K
  const int a_r  = tid >> 3;                      // 0..63
  const int b_c4 = tid & 31;                      // float4 col in 128-wide N
  const int b_r  = tid >> 5;                      // 0..15

  v4f areg[4], breg[2];

  auto load_tiles = [&](int kb) {
    const int k0 = kb * TK;
    #pragma unroll
    for (int rr = 0; rr < 4; ++rr)
      areg[rr] = *(const v4f*)(X + (size_t)(m0 + a_r + rr * 64) * Dn + k0 + a_c4 * 4);
    #pragma unroll
    for (int rr = 0; rr < 2; ++rr)
      breg[rr] = *(const v4f*)(W + (size_t)(k0 + b_r + rr * 16) * Un + n0 + b_c4 * 4);
  };
  auto store_tiles = [&](int buf) {
    #pragma unroll
    for (int rr = 0; rr < 4; ++rr) {
      __bf16* dst = &As[buf][(a_r + rr * 64) * LDA + a_c4 * 4];
      dst[0] = (__bf16)areg[rr].x; dst[1] = (__bf16)areg[rr].y;
      dst[2] = (__bf16)areg[rr].z; dst[3] = (__bf16)areg[rr].w;
    }
    #pragma unroll
    for (int rr = 0; rr < 2; ++rr) {
      const int kr = b_r + rr * 16;
      Bs[buf][(b_c4 * 4 + 0) * LDA + kr] = (__bf16)breg[rr].x;
      Bs[buf][(b_c4 * 4 + 1) * LDA + kr] = (__bf16)breg[rr].y;
      Bs[buf][(b_c4 * 4 + 2) * LDA + kr] = (__bf16)breg[rr].z;
      Bs[buf][(b_c4 * 4 + 3) * LDA + kr] = (__bf16)breg[rr].w;
    }
  };

  load_tiles(0);
  store_tiles(0);
  __syncthreads();

  for (int kb = 0; kb < NKB; ++kb) {
    const int cb = kb & 1;
    if (kb + 1 < NKB) {
      load_tiles(kb + 1);                         // global loads overlap WMMA below
      __builtin_prefetch(X + (size_t)(m0 + a_r) * Dn + (kb + 2) * TK, 0, 3);
    }

    // fragments from LDS per ISA 7.12.2 layouts
    V16 af[2], bf[4];
    #pragma unroll
    for (int i = 0; i < 2; ++i) {                 // A: lane = m + 16*hi
      const __bf16* ap = &As[cb][(wm + i * 16 + nl) * LDA + 8 * hid];
      af[i].h[0] = *(const v8bf*)ap;              // e 0..7  : K = 8*hi + e
      af[i].h[1] = *(const v8bf*)(ap + 16);       // e 8..15 : K = 8 + 8*hi + e
    }
    #pragma unroll
    for (int j = 0; j < 4; ++j) {                 // B: same K order as A
      const __bf16* bp = &Bs[cb][(wn + j * 16 + nl) * LDA + 8 * hid];
      bf[j].h[0] = *(const v8bf*)bp;
      bf[j].h[1] = *(const v8bf*)(bp + 16);
    }
    #pragma unroll
    for (int i = 0; i < 2; ++i)
      #pragma unroll
      for (int j = 0; j < 4; ++j)
        acc[i][j] = __builtin_amdgcn_wmma_f32_16x16x32_bf16(
            false, af[i].v, false, bf[j].v, (short)0, acc[i][j], false, false);

    if (kb + 1 < NKB) store_tiles(cb ^ 1);        // write other buffer: no race
    __syncthreads();
  }

  // epilogue: + bias, store f32 xw (C/D layout: M = r + 8*hi, N = nl)
  #pragma unroll
  for (int j = 0; j < 4; ++j) {
    const int n = n0 + wn + j * 16 + nl;
    const float bv = bias[n];
    #pragma unroll
    for (int i = 0; i < 2; ++i)
      #pragma unroll
      for (int r = 0; r < 8; ++r) {
        const int m = m0 + wm + i * 16 + r + 8 * hid;
        out[(size_t)m * Un + n] = acc[i][j][r] + bv;
      }
  }
}

// ---------------------------------------------------------------------------
// Phase 2: sequential scan  h = tanh(xw_t + h @ R)
// 4 batch-groups (16 rows) x 32 col-chunk WGs (32 cols), 2 waves per WG.
// R chunk in LDS (B-fragment layout). Per step, h fragments are pulled
// L2 -> LDS with GLOBAL_LOAD_ASYNC_TO_LDS_B128 (ASYNCcnt) and shared by
// both waves. Device-scope atomic barrier + agent fences per step.
// ---------------------------------------------------------------------------
static constexpr int GROUPS = 4, CHUNKS = 32, NW = 2, KC = Un / 32;   // KC = 32
static constexpr int HFRAG = KC * 32 * 16;        // bf16 elems per group buffer (32 KB)

__device__ __forceinline__ void group_barrier(unsigned int* bar, int g,
                                              unsigned int round) {
  __builtin_amdgcn_fence(__ATOMIC_RELEASE, "agent");
  __syncthreads();
  if (threadIdx.x == 0) {
    __hip_atomic_fetch_add(&bar[g], 1u, __ATOMIC_RELAXED, __HIP_MEMORY_SCOPE_AGENT);
    while (__hip_atomic_load(&bar[g], __ATOMIC_RELAXED, __HIP_MEMORY_SCOPE_AGENT)
           < round * (unsigned)CHUNKS)
      __builtin_amdgcn_s_sleep(1);
  }
  __syncthreads();
  __builtin_amdgcn_fence(__ATOMIC_ACQUIRE, "agent");
}

__global__ __launch_bounds__(64)
void rnn_scan(const float* __restrict__ R, const float* __restrict__ h0,
              float* __restrict__ out,           // holds xw, overwritten with h
              __bf16* __restrict__ hbuf,         // [2][GROUPS][HFRAG]
              unsigned int* __restrict__ bar) {
  __shared__ __align__(16) __bf16 Rlds[KC * NW * 32 * 16];   // 64 KB
  __shared__ __align__(16) __bf16 Hs[HFRAG];                 // 32 KB h stage

  const int g    = blockIdx.x / CHUNKS;
  const int c    = blockIdx.x % CHUNKS;
  const int tid  = threadIdx.x;
  const int lane = tid & 31;
  const int w    = tid >> 5;                      // wave = n-tile (0..1)
  const int nl   = lane & 15;
  const int hid  = lane >> 4;

  // 1) stage R[:, c*32 .. +32] into LDS, pre-swizzled into B-fragment order
  for (int idx = tid; idx < KC * NW * 32 * 16; idx += 64) {
    const int e  = idx & 15;
    const int ln = (idx >> 4) & 31;
    const int ww = (idx >> 9) & (NW - 1);
    const int kc = idx >> 10;
    const int k  = kc * 32 + 8 * (ln >> 4) + (e & 7) + ((e >> 3) << 4);  // B: K = 8*hi + e (e<8), 16 + 8*hi + e-8
    const int n  = c * 32 + ww * 16 + (ln & 15);
    Rlds[idx] = (__bf16)R[(size_t)k * Un + n];
  }

  __bf16* buf0 = hbuf + (size_t)g * HFRAG;
  __bf16* buf1 = hbuf + (size_t)(GROUPS + g) * HFRAG;

  // 2) chunk-0 WG scatters h0 into A-fragment layout of buffer 0
  if (c == 0) {
    for (int idx = tid; idx < 16 * Un; idx += 64) {
      const int m  = idx >> 10;                   // batch row in group
      const int u  = idx & (Un - 1);
      const int kc = u >> 5, kk = u & 31, q = kk >> 3;
      const int hi = q & 1;
      const int e  = (kk & 7) + ((q >> 1) << 3);
      buf0[(kc * 32 + hi * 16 + m) * 16 + e] =
          (__bf16)h0[(size_t)(g * 16 + m) * Un + u];
    }
  }

  // async L2 -> LDS staging of the 32 KB h-fragment buffer (shared by waves)
  auto stage_h = [&](const __bf16* cur) {
#if defined(__gfx1250__) && __has_builtin(__builtin_amdgcn_global_load_async_to_lds_b128)
    for (int i = w; i < (HFRAG * 2) / 512; i += NW) {   // 64 chunks of 512 B
      const __bf16* gsrc = cur + i * 256 + lane * 8;    // 16 B per lane
      __bf16*       ldst = Hs  + i * 256 + lane * 8;
      __builtin_amdgcn_global_load_async_to_lds_b128(
          (gv4i*)(unsigned long long)gsrc,
          (lv4i*)(unsigned)(unsigned long long)ldst,
          0, 0);
    }
#if __has_builtin(__builtin_amdgcn_s_wait_asynccnt)
    __builtin_amdgcn_s_wait_asynccnt(0);
#else
    asm volatile("s_wait_asynccnt 0x0" ::: "memory");
#endif
#else
    for (int i = tid; i < HFRAG / 8; i += 64)
      *(v8bf*)(Hs + i * 8) = *(const v8bf*)(cur + i * 8);
#endif
    __syncthreads();
  };

  unsigned int round = 1;
  group_barrier(bar, g, round++);

  // this thread's output column (fixed) -> scatter target in A-frag layout
  const int u   = c * 32 + w * 16 + nl;
  const int kct = u >> 5, kk = u & 31, q = kk >> 3;
  const int hit = q & 1;
  const int et  = (kk & 7) + ((q >> 1) << 3);

  for (int t = 0; t < Tn; ++t) {
    const __bf16* cur = (t & 1) ? buf1 : buf0;
    __bf16*       nxt = (t & 1) ? buf0 : buf1;

    stage_h(cur);

    v8f acc = {};
    #pragma unroll 4
    for (int kc = 0; kc < KC; ++kc) {
      V16 av, bv;
      const __bf16* ap = Hs + (size_t)(kc * 32 + lane) * 16;
      av.h[0] = *(const v8bf*)ap;
      av.h[1] = *(const v8bf*)(ap + 8);
      const __bf16* bp = Rlds + ((kc * NW + w) * 32 + lane) * 16;
      bv.h[0] = *(const v8bf*)bp;
      bv.h[1] = *(const v8bf*)(bp + 8);
      acc = __builtin_amdgcn_wmma_f32_16x16x32_bf16(
          false, av.v, false, bv.v, (short)0, acc, false, false);
    }

    #pragma unroll
    for (int r = 0; r < 8; ++r) {
      const int m = r + 8 * hid;                  // batch row within group
      const size_t oidx = ((size_t)(g * 16 + m) * Tn + t) * (size_t)Un + u;
      const float hv = tanhf(acc[r] + out[oidx]); // xw_t read in-place
      out[oidx] = hv;                             // h_t written in-place
      nxt[(kct * 32 + hit * 16 + m) * 16 + et] = (__bf16)hv;
    }

    group_barrier(bar, g, round++);
  }
}

// ---------------------------------------------------------------------------
extern "C" void kernel_launch(void* const* d_in, const int* in_sizes, int n_in,
                              void* d_out, int out_size, void* d_ws, size_t ws_size,
                              hipStream_t stream) {
  (void)in_sizes; (void)n_in; (void)out_size; (void)ws_size;
  const float* x    = (const float*)d_in[0];
  const float* h0   = (const float*)d_in[1];
  const float* W    = (const float*)d_in[2];
  const float* R    = (const float*)d_in[3];
  const float* bias = (const float*)d_in[4];
  float* out = (float*)d_out;

  unsigned int* bar = (unsigned int*)d_ws;                 // 4 counters
  __bf16* hbuf = (__bf16*)((char*)d_ws + 256);             // 256 KB double buffer

  (void)hipMemsetAsync(d_ws, 0, 256, stream);              // reset barrier counters

  dim3 g1(Un / TN, Mtot / TM);                             // (8, 128)
  xw_gemm<<<g1, 512, 0, stream>>>(x, W, bias, out);

  rnn_scan<<<GROUPS * CHUNKS, 64, 0, stream>>>(R, h0, out, hbuf, bar);
}
